// ReorderLlamaAttention_10514079940701
// MI455X (gfx1250) — compile-verified, matrix operand order checked by reading, not run on hardware
//
#include <hip/hip_runtime.h>
#include <hip/hip_bf16.h>
#include <cstdint>

// Problem constants (B=1)
static constexpr int S_LEN = 2048;
static constexpr int HDIM  = 4096;
static constexpr int NH    = 32;
static constexpr int NKV   = 8;
static constexpr int HD    = 128;

typedef __attribute__((ext_vector_type(16))) __bf16 v16bf;
typedef __attribute__((ext_vector_type(8)))  float  v8f;

union FragBF { v16bf v; uint4 u[2]; };

// Async copy of 16 bytes global -> LDS (ASYNCcnt-tracked, no VGPR round trip).
__device__ __forceinline__ void async_copy_b128(void* lds_ptr, const void* gptr) {
  unsigned lds_addr = (unsigned)(size_t)lds_ptr;   // low 32 bits = LDS offset
  asm volatile("global_load_async_to_lds_b128 %0, %1, off"
               :: "v"(lds_addr), "v"(gptr) : "memory");
}

__device__ __forceinline__ void async_wait_all() {
  asm volatile("s_wait_asynccnt 0" ::: "memory");
}

// Load one bf16 WMMA fragment from LDS (row-major tile, `ld` element stride).
// Lane half=l>>4 selects K in {half*8..+7} and {16+half*8..+7}: 2x16B chunks.
__device__ __forceinline__ v16bf load_frag(const __bf16* p, int row, int ld, int half) {
  FragBF f;
  f.u[0] = *reinterpret_cast<const uint4*>(p + row * ld + half * 8);
  f.u[1] = *reinterpret_cast<const uint4*>(p + row * ld + 16 + half * 8);
  return f.v;
}

// ---------------------------------------------------------------------------
// Generic C(f32, MxN) = A(bf16, MxK, lda) x W(bf16, NxK, ldw)^T
// 128x128 block tile, 8 waves; wave tile 32x64 (2x4 WMMA tiles).
// Tiles staged with global_load_async_to_lds_b128.
// grid = (N/128, M/128). Used for QKV + output projections.
// ---------------------------------------------------------------------------
__global__ __launch_bounds__(256) void gemm_bf16_nt(
    const __bf16* __restrict__ A, const __bf16* __restrict__ W,
    float* __restrict__ C, int K, int lda, int ldw, int ldc) {
  __shared__ __attribute__((aligned(16))) __bf16 sA[128 * 32];
  __shared__ __attribute__((aligned(16))) __bf16 sB[128 * 32];
  const int bx = blockIdx.x;              // N tile
  const int by = blockIdx.y;              // M tile
  const int t    = threadIdx.x;
  const int w    = t >> 5;
  const int l    = t & 31;
  const int half = l >> 4;
  const int idx  = l & 15;
  const int wm   = (w & 3) * 32;
  const int wn   = (w >> 2) * 64;

  v8f acc[2][4];
#pragma unroll
  for (int i = 0; i < 2; ++i)
#pragma unroll
    for (int j = 0; j < 4; ++j)
      acc[i][j] = (v8f){0.f, 0.f, 0.f, 0.f, 0.f, 0.f, 0.f, 0.f};

  for (int k = 0; k < K; k += 32) {
    // Async stage A and B 128x32 tiles straight into LDS.
#pragma unroll
    for (int it = 0; it < 2; ++it) {
      int cid = t + it * 256;
      int row = cid >> 2;
      int cc  = (cid & 3) * 8;
      async_copy_b128(sA + row * 32 + cc,
                      A + (size_t)(by * 128 + row) * lda + k + cc);
      async_copy_b128(sB + row * 32 + cc,
                      W + (size_t)(bx * 128 + row) * ldw + k + cc);
    }
    async_wait_all();
    __syncthreads();

    v16bf af[2], bfrag[4];
#pragma unroll
    for (int mt = 0; mt < 2; ++mt) af[mt]    = load_frag(sA, wm + mt * 16 + idx, 32, half);
#pragma unroll
    for (int nt = 0; nt < 4; ++nt) bfrag[nt] = load_frag(sB, wn + nt * 16 + idx, 32, half);

#pragma unroll
    for (int mt = 0; mt < 2; ++mt)
#pragma unroll
      for (int nt = 0; nt < 4; ++nt)
        acc[mt][nt] = __builtin_amdgcn_wmma_f32_16x16x32_bf16(
            false, af[mt], false, bfrag[nt], (short)0, acc[mt][nt], false, false);
    __syncthreads();
  }

#pragma unroll
  for (int mt = 0; mt < 2; ++mt) {
#pragma unroll
    for (int nt = 0; nt < 4; ++nt) {
      int col   = bx * 128 + wn + nt * 16 + idx;
      int rbase = by * 128 + wm + mt * 16 + half * 8;
#pragma unroll
      for (int r = 0; r < 8; ++r)
        C[(size_t)(rbase + r) * ldc + col] = acc[mt][nt][r];
    }
  }
}

// ---------------------------------------------------------------------------
// Fused flash attention. grid = (S/128, NH), 256 threads = 8 waves.
// Each wave owns 16 q-rows; online softmax in fp32; scores and P*V via WMMA.
// Q is pre-scaled by 1/sqrt(HD) (folded into RoPE).
// K/Vt blocks staged with async-to-LDS; output written as bf16 into Ab.
// ---------------------------------------------------------------------------
__global__ __launch_bounds__(256) void flash_attn_kernel(
    const __bf16* __restrict__ Qb,   // (S, NH*HD), pre-scaled
    const __bf16* __restrict__ Kb,   // (S, NKV*HD)
    const __bf16* __restrict__ Vtb,  // (NKV, HD, S)
    __bf16* __restrict__ Ab) {       // (S, NH*HD)
  __shared__ __attribute__((aligned(16))) __bf16 sK[128 * 128];      // 32 KB: j x d
  __shared__ __attribute__((aligned(16))) __bf16 sV[128 * 128];      // 32 KB: d x j
  __shared__ __attribute__((aligned(16))) __bf16 sP[8 * 16 * 128];   // 32 KB: wave-private P
  const int qb = blockIdx.x;
  const int h  = blockIdx.y;
  const int kv = h >> 2;                 // NH/NKV = 4
  const int t = threadIdx.x;
  const int w = t >> 5, l = t & 31, half = l >> 4, idx = l & 15;

  // Q fragments for this wave's 16 rows, kept in registers (HD=128 -> 4 frags)
  FragBF qf[4];
  {
    const __bf16* qp = Qb + (size_t)(qb * 128 + w * 16 + idx) * (NH * HD) + h * HD;
#pragma unroll
    for (int c = 0; c < 4; ++c) {
      qf[c].u[0] = *reinterpret_cast<const uint4*>(qp + c * 32 + half * 8);
      qf[c].u[1] = *reinterpret_cast<const uint4*>(qp + c * 32 + 16 + half * 8);
    }
  }

  v8f o[8];
#pragma unroll
  for (int nt = 0; nt < 8; ++nt) o[nt] = (v8f){0.f, 0.f, 0.f, 0.f, 0.f, 0.f, 0.f, 0.f};
  float rm[8], rs[8];
#pragma unroll
  for (int r = 0; r < 8; ++r) { rm[r] = -1e30f; rs[r] = 0.f; }

  __bf16* myP = sP + w * (16 * 128);

  for (int kb = 0; kb <= qb; ++kb) {
    __syncthreads();
    // Async stage K block (128 j x 128 d) and Vt block (128 d x 128 j).
#pragma unroll
    for (int it = 0; it < 8; ++it) {
      int cid = t + it * 256;          // 2048 chunks of 8 bf16 per tile
      int row = cid >> 4;
      int cc  = (cid & 15) * 8;
      async_copy_b128(sK + row * 128 + cc,
                      Kb + (size_t)(kb * 128 + row) * (NKV * HD) + kv * HD + cc);
      async_copy_b128(sV + row * 128 + cc,
                      Vtb + (size_t)kv * HD * S_LEN + (size_t)row * S_LEN + kb * 128 + cc);
    }
    async_wait_all();
    __syncthreads();

    // scores: wave computes 16 x 128 tile = 8 accumulators
    v8f s[8];
#pragma unroll
    for (int nt = 0; nt < 8; ++nt) s[nt] = (v8f){0.f, 0.f, 0.f, 0.f, 0.f, 0.f, 0.f, 0.f};
#pragma unroll
    for (int c = 0; c < 4; ++c) {
#pragma unroll
      for (int nt = 0; nt < 8; ++nt) {
        FragBF kfr;
        kfr.u[0] = *reinterpret_cast<const uint4*>(sK + (nt * 16 + idx) * 128 + c * 32 + half * 8);
        kfr.u[1] = *reinterpret_cast<const uint4*>(sK + (nt * 16 + idx) * 128 + c * 32 + 16 + half * 8);
        s[nt] = __builtin_amdgcn_wmma_f32_16x16x32_bf16(
            false, qf[c].v, false, kfr.v, (short)0, s[nt], false, false);
      }
    }

    // causal mask on the diagonal block
    if (kb == qb) {
#pragma unroll
      for (int nt = 0; nt < 8; ++nt) {
        int col = nt * 16 + idx;
#pragma unroll
        for (int r = 0; r < 8; ++r) {
          int row = w * 16 + half * 8 + r;
          if (col > row) s[nt][r] = -1e30f;
        }
      }
    }

    // online softmax: rows r+8*half; reductions across the 16 lanes of a half
#pragma unroll
    for (int r = 0; r < 8; ++r) {
      float bm = s[0][r];
#pragma unroll
      for (int nt = 1; nt < 8; ++nt) bm = fmaxf(bm, s[nt][r]);
      bm = fmaxf(bm, __shfl_xor(bm, 1, 32));
      bm = fmaxf(bm, __shfl_xor(bm, 2, 32));
      bm = fmaxf(bm, __shfl_xor(bm, 4, 32));
      bm = fmaxf(bm, __shfl_xor(bm, 8, 32));
      float nm = fmaxf(rm[r], bm);
      float al = __expf(rm[r] - nm);
      rm[r] = nm;
      float ps = 0.f;
#pragma unroll
      for (int nt = 0; nt < 8; ++nt) {
        float p = __expf(s[nt][r] - nm);
        s[nt][r] = p;
        ps += p;
      }
      ps += __shfl_xor(ps, 1, 32);
      ps += __shfl_xor(ps, 2, 32);
      ps += __shfl_xor(ps, 4, 32);
      ps += __shfl_xor(ps, 8, 32);
      rs[r] = rs[r] * al + ps;
#pragma unroll
      for (int nt = 0; nt < 8; ++nt) o[nt][r] *= al;
    }

    // write P tile (C-layout) to wave-private LDS, re-read as A-fragments
#pragma unroll
    for (int nt = 0; nt < 8; ++nt)
#pragma unroll
      for (int r = 0; r < 8; ++r)
        myP[(half * 8 + r) * 128 + nt * 16 + idx] = (__bf16)s[nt][r];

    // o += P (16x128) x V (128x128): B = Vt rows (d), K-pattern over j
#pragma unroll
    for (int c = 0; c < 4; ++c) {
      FragBF pf;
      pf.u[0] = *reinterpret_cast<const uint4*>(myP + idx * 128 + c * 32 + half * 8);
      pf.u[1] = *reinterpret_cast<const uint4*>(myP + idx * 128 + c * 32 + 16 + half * 8);
#pragma unroll
      for (int nt = 0; nt < 8; ++nt) {
        FragBF vf;
        vf.u[0] = *reinterpret_cast<const uint4*>(sV + (nt * 16 + idx) * 128 + c * 32 + half * 8);
        vf.u[1] = *reinterpret_cast<const uint4*>(sV + (nt * 16 + idx) * 128 + c * 32 + 16 + half * 8);
        o[nt] = __builtin_amdgcn_wmma_f32_16x16x32_bf16(
            false, pf.v, false, vf.v, (short)0, o[nt], false, false);
      }
    }
  }

  // normalize and store bf16 attention output
#pragma unroll
  for (int nt = 0; nt < 8; ++nt) {
    int col = h * HD + nt * 16 + idx;
#pragma unroll
    for (int r = 0; r < 8; ++r) {
      int row = qb * 128 + w * 16 + half * 8 + r;
      Ab[(size_t)row * (NH * HD) + col] = (__bf16)(o[nt][r] / rs[r]);
    }
  }
}

// ---------------------------------------------------------------------------
// Elementwise helpers
// ---------------------------------------------------------------------------
__global__ void scatter_inv_kernel(const int* __restrict__ p, int* __restrict__ inv, int n) {
  int i = blockIdx.x * blockDim.x + threadIdx.x;
  if (i < n) inv[p[i]] = i;
}

__global__ void f32_to_bf16_kernel(const float* __restrict__ x, __bf16* __restrict__ y, int n) {
  int i = blockIdx.x * blockDim.x + threadIdx.x;
  if (i < n) y[i] = (__bf16)x[i];
}

__global__ void permute_weight_kernel(const float* __restrict__ W, const int* __restrict__ rowp,
                                      const int* __restrict__ invc, __bf16* __restrict__ dst,
                                      int rows, int cols) {
  int i = blockIdx.x * blockDim.x + threadIdx.x;
  int n = rows * cols;
  if (i >= n) return;
  int r = i / cols, c = i - r * cols;
  dst[i] = (__bf16)W[(size_t)rowp[r] * cols + invc[c]];
}

// RoPE on fp32 activations (s, h, d) with row width `width`, emitting bf16,
// optionally pre-scaling (1/sqrt(HD) for Q).
__global__ void rope_kernel(const float* __restrict__ X, __bf16* __restrict__ Y,
                            const float* __restrict__ cosb, const float* __restrict__ sinb,
                            int width, int total, float scale) {
  int i = blockIdx.x * blockDim.x + threadIdx.x;
  if (i >= total) return;
  int d = i & (HD - 1);
  int s = i / width;
  float c  = cosb[s * HD + d];
  float sn = sinb[s * HD + d];
  float x  = X[i];
  float o  = (d < HD / 2) ? -X[i + HD / 2] : X[i - HD / 2];
  Y[i] = (__bf16)((x * c + o * sn) * scale);
}

__global__ void transpose_v_kernel(const float* __restrict__ V, __bf16* __restrict__ Vt) {
  int i = blockIdx.x * blockDim.x + threadIdx.x;
  int n = NKV * HD * S_LEN;
  if (i >= n) return;
  int s  = i & (S_LEN - 1);
  int d  = (i >> 11) & (HD - 1);
  int kv = i >> 18;
  Vt[i] = (__bf16)V[(size_t)s * (NKV * HD) + kv * HD + d];
}

extern "C" void kernel_launch(void* const* d_in, const int* in_sizes, int n_in,
                              void* d_out, int out_size, void* d_ws, size_t ws_size,
                              hipStream_t stream) {
  const float* hidden = (const float*)d_in[0];
  const float* Wq     = (const float*)d_in[1];
  const float* Wk     = (const float*)d_in[2];
  const float* Wv     = (const float*)d_in[3];
  const float* Wo     = (const float*)d_in[4];
  const float* cosb   = (const float*)d_in[5];
  const float* sinb   = (const float*)d_in[6];
  // d_in[7] = attention_mask (causal; baked into flash kernel)
  const int* qc = (const int*)d_in[8];
  const int* qr = (const int*)d_in[9];
  const int* kc = (const int*)d_in[10];
  const int* kr = (const int*)d_in[11];
  const int* vc = (const int*)d_in[12];
  const int* vr = (const int*)d_in[13];
  const int* oc = (const int*)d_in[14];
  const int* orow = (const int*)d_in[15];
  float* out = (float*)d_out;

  char* ws = (char*)d_ws;
  size_t off = 0;
  auto alloc = [&](size_t bytes) -> char* {
    char* p = ws + off;
    off += (bytes + 255) & ~(size_t)255;
    return p;
  };

  int* inv_qc = (int*)alloc(HDIM * sizeof(int));
  int* inv_kc = (int*)alloc(HDIM * sizeof(int));
  int* inv_vc = (int*)alloc(HDIM * sizeof(int));
  int* inv_oc = (int*)alloc((size_t)NH * HD * sizeof(int));

  __bf16* Xb  = (__bf16*)alloc((size_t)S_LEN * HDIM * 2);
  __bf16* Wqp = (__bf16*)alloc((size_t)NH * HD * HDIM * 2);
  __bf16* Wkp = (__bf16*)alloc((size_t)NKV * HD * HDIM * 2);
  __bf16* Wvp = (__bf16*)alloc((size_t)NKV * HD * HDIM * 2);
  __bf16* Wop = (__bf16*)alloc((size_t)HDIM * NH * HD * 2);

  float* Qf = (float*)alloc((size_t)S_LEN * NH * HD * 4);
  float* Kf = (float*)alloc((size_t)S_LEN * NKV * HD * 4);
  float* Vf = (float*)alloc((size_t)S_LEN * NKV * HD * 4);

  __bf16* Qb  = (__bf16*)alloc((size_t)S_LEN * NH * HD * 2);
  __bf16* Kb  = (__bf16*)alloc((size_t)S_LEN * NKV * HD * 2);
  __bf16* Vtb = (__bf16*)alloc((size_t)NKV * HD * S_LEN * 2);
  __bf16* Ab  = (__bf16*)alloc((size_t)S_LEN * NH * HD * 2);

  auto cdiv = [](int a, int b) { return (a + b - 1) / b; };

  // 1) inverse column permutations
  scatter_inv_kernel<<<cdiv(HDIM, 256), 256, 0, stream>>>(qc, inv_qc, HDIM);
  scatter_inv_kernel<<<cdiv(HDIM, 256), 256, 0, stream>>>(kc, inv_kc, HDIM);
  scatter_inv_kernel<<<cdiv(HDIM, 256), 256, 0, stream>>>(vc, inv_vc, HDIM);
  scatter_inv_kernel<<<cdiv(NH * HD, 256), 256, 0, stream>>>(oc, inv_oc, NH * HD);

  // 2) hidden -> bf16; permutation-folded bf16 weights
  f32_to_bf16_kernel<<<cdiv(S_LEN * HDIM, 256), 256, 0, stream>>>(hidden, Xb, S_LEN * HDIM);
  permute_weight_kernel<<<cdiv(NH * HD * HDIM, 256), 256, 0, stream>>>(Wq, qr, inv_qc, Wqp, NH * HD, HDIM);
  permute_weight_kernel<<<cdiv(NKV * HD * HDIM, 256), 256, 0, stream>>>(Wk, kr, inv_kc, Wkp, NKV * HD, HDIM);
  permute_weight_kernel<<<cdiv(NKV * HD * HDIM, 256), 256, 0, stream>>>(Wv, vr, inv_vc, Wvp, NKV * HD, HDIM);
  permute_weight_kernel<<<cdiv(HDIM * NH * HD, 256), 256, 0, stream>>>(Wo, orow, inv_oc, Wop, HDIM, NH * HD);

  // 3) QKV projections (WMMA bf16, f32 accum)
  gemm_bf16_nt<<<dim3((NH * HD) / 128, S_LEN / 128), 256, 0, stream>>>(
      Xb, Wqp, Qf, HDIM, HDIM, HDIM, NH * HD);
  gemm_bf16_nt<<<dim3((NKV * HD) / 128, S_LEN / 128), 256, 0, stream>>>(
      Xb, Wkp, Kf, HDIM, HDIM, HDIM, NKV * HD);
  gemm_bf16_nt<<<dim3((NKV * HD) / 128, S_LEN / 128), 256, 0, stream>>>(
      Xb, Wvp, Vf, HDIM, HDIM, HDIM, NKV * HD);

  // 4) RoPE (Q gets 1/sqrt(HD) folded in) and V transpose
  rope_kernel<<<cdiv(S_LEN * NH * HD, 256), 256, 0, stream>>>(
      Qf, Qb, cosb, sinb, NH * HD, S_LEN * NH * HD, 0.08838834764831845f);
  rope_kernel<<<cdiv(S_LEN * NKV * HD, 256), 256, 0, stream>>>(
      Kf, Kb, cosb, sinb, NKV * HD, S_LEN * NKV * HD, 1.0f);
  transpose_v_kernel<<<cdiv(NKV * HD * S_LEN, 256), 256, 0, stream>>>(Vf, Vtb);

  // 5) fused flash attention: all heads + q-blocks in one launch
  flash_attn_kernel<<<dim3(S_LEN / 128, NH), 256, 0, stream>>>(Qb, Kb, Vtb, Ab);

  // 6) output projection
  gemm_bf16_nt<<<dim3(HDIM / 128, S_LEN / 128), 256, 0, stream>>>(
      Ab, Wop, out, NH * HD, NH * HD, NH * HD, HDIM);
}
